// ggm_20933670601167
// MI455X (gfx1250) — compile-verified, matrix operand order checked by reading, not run on hardware
//
#include <hip/hip_runtime.h>
#include <math.h>

// ---------------------------------------------------------------------------
// GGM graph-GRU encoder for MI455X (gfx1250), wave32 + WMMA bf16.
// ---------------------------------------------------------------------------

typedef __attribute__((ext_vector_type(16))) __bf16 v16bf;
typedef __attribute__((ext_vector_type(8)))  float  v8f;
typedef __attribute__((ext_vector_type(4)))  float  v4f;

#define WMMA_BF16(a, b, c) \
  __builtin_amdgcn_wmma_f32_16x16x32_bf16(false, (a), false, (b), (short)0, (c), false, false)

__device__ __forceinline__ void atomic_add_f32(float* p, float v) {
  // no-return f32 atomic (STOREcnt path) -- avoid CAS-loop fallback
  asm volatile("global_atomic_add_f32 %0, %1, off" :: "v"(p), "v"(v) : "memory");
}

__device__ __forceinline__ unsigned short f2bf_bits(float f) {
  union { __bf16 b; unsigned short u; } c;
  c.b = (__bf16)f;
  return c.u;
}

// A fragment (16x32 bf16) built in-register from a row of a f32 [*,128] matrix.
// ISA 7.12.2: lane<16 -> K = kb..kb+7 and kb+16..kb+23 with kb = 8*(lane>=16).
__device__ __forceinline__ v16bf a_frag_f32(const float* __restrict__ rowptr,
                                            int kt, int lane, float scale) {
  int kb = kt * 32 + ((lane >> 4) << 3);
  v16bf a;
#pragma unroll
  for (int i = 0; i < 8; i++) a[i] = (__bf16)(rowptr[kb + i] * scale);
#pragma unroll
  for (int i = 0; i < 8; i++) a[8 + i] = (__bf16)(rowptr[kb + 16 + i] * scale);
  return a;
}

// B fragment: pre-packed lane-contiguous 32B chunks (see k_prepack).
__device__ __forceinline__ v16bf b_frag(const unsigned short* __restrict__ packed,
                                        int ntiles, int kt, int nt, int lane) {
  return *(const v16bf*)(packed + (((kt * ntiles) + nt) * 32 + lane) * 16);
}

// ---------------------------------------------------------------------------
// Pre-pack a f32 [128 x ncols] weight into bf16 WMMA B fragments.
// dst[t], t = ((kt*ntiles+nt)*32 + lane)*16 + i   maps to W[k][n] with
// k = kt*32 + 8*(lane>>4) + (i<8 ? i : i+8), n = nt*16 + (lane&15).
// ---------------------------------------------------------------------------
__global__ void k_prepack(const float* __restrict__ src,
                          unsigned short* __restrict__ dst, int ncols) {
  int t = blockIdx.x * blockDim.x + threadIdx.x;
  int total = 128 * ncols;
  if (t >= total) return;
  int ntiles = ncols >> 4;
  int i  = t & 15;
  int l  = (t >> 4) & 31;
  int f  = t >> 9;
  int nt = f % ntiles;
  int kt = f / ntiles;
  int k = kt * 32 + ((l >> 4) << 3) + (i < 8 ? i : i + 8);
  int n = nt * 16 + (l & 15);
  dst[t] = f2bf_bits(src[k * ncols + n]);
}

__global__ void k_zero(float* __restrict__ p, long n) {
  long t = (long)blockIdx.x * blockDim.x + threadIdx.x;
  if (t < n) p[t] = 0.f;
}

// cond_term[k] = condition @ Wc_k + enc_U_b[k]   (3 x 128)
__global__ void k_cond(const float* __restrict__ cond,
                       const float* __restrict__ encUW,
                       const float* __restrict__ encUb,
                       float* __restrict__ out) {
  int k = blockIdx.x;
  int c = threadIdx.x;
  const float* Wc = encUW + (long)k * 388 * 128 + 384 * 128;
  float s = encUb[k * 128 + c];
#pragma unroll
  for (int j = 0; j < 4; j++) s += cond[j] * Wc[j * 128 + c];
  out[k * 128 + c] = s;
}

// h = node_feats @ W_node_emb   (N x 14)(14 x 128) -- tiny K, plain VALU
__global__ void k_node_embed(const float* __restrict__ nf,
                             const float* __restrict__ W,
                             float* __restrict__ h, long N) {
  long t = (long)blockIdx.x * blockDim.x + threadIdx.x;
  if (t >= N * 128) return;
  long row = t >> 7;
  int  c   = (int)(t & 127);
  const float* f = nf + row * 14;
  float s = 0.f;
#pragma unroll
  for (int j = 0; j < 14; j++) s += f[j] * W[j * 128 + c];
  h[t] = s;
}

// Ehat[dst] += edge_feats @ W_edge_emb  (once; round-invariant), plus degree.
__global__ void k_edge_embed_scatter(const float* __restrict__ ef,
                                     const float* __restrict__ W,
                                     const int* __restrict__ edst,
                                     float* __restrict__ ehat,
                                     float* __restrict__ deg, long E) {
  long t = (long)blockIdx.x * blockDim.x + threadIdx.x;
  if (t >= E * 32) return;
  long e = t >> 5;
  int  q = (int)(t & 31);
  const float* f = ef + e * 11;
  float fv[11];
#pragma unroll
  for (int j = 0; j < 11; j++) fv[j] = f[j];
  int d = edst[e];
  float* out = ehat + (long)d * 128 + q * 4;
#pragma unroll
  for (int c = 0; c < 4; c++) {
    float s = 0.f;
#pragma unroll
    for (int j = 0; j < 11; j++) s += fv[j] * W[j * 128 + q * 4 + c];
    atomic_add_f32(out + c, s);
  }
  if (q == 0) atomic_add_f32(deg + d, 1.0f);
}

// agg[dst] += h[src]  (per round; h is 25.6MB -> L2-resident gathers)
__global__ void k_scatter_h(const float* __restrict__ h,
                            const int* __restrict__ esrc,
                            const int* __restrict__ edst,
                            float* __restrict__ agg, long E) {
  long t = (long)blockIdx.x * blockDim.x + threadIdx.x;
  if (t >= E * 32) return;
  long e = t >> 5;
  int  q = (int)(t & 31);
  int s = esrc[e], d = edst[e];
  v4f v = *(const v4f*)(h + (long)s * 128 + q * 4);
  float* out = agg + (long)d * 128 + q * 4;
  atomic_add_f32(out + 0, v[0]);
  atomic_add_f32(out + 1, v[1]);
  atomic_add_f32(out + 2, v[2]);
  atomic_add_f32(out + 3, v[3]);
}

// m = (deg .* h) @ Wv + agg @ Ww + Ehat @ We + deg .* cond_term
// 16 rows/block, 8 waves = 8 column tiles; 12 WMMAs per wave.
__global__ void __launch_bounds__(256) k_gemm_msg(
    const float* __restrict__ h, const float* __restrict__ agg,
    const float* __restrict__ ehat, const float* __restrict__ deg,
    const float* __restrict__ cond,
    const unsigned short* __restrict__ pWv,
    const unsigned short* __restrict__ pWw,
    const unsigned short* __restrict__ pWe,
    float* __restrict__ m, int N) {
  int lane = threadIdx.x & 31;
  int nt   = threadIdx.x >> 5;
  int row0 = blockIdx.x << 4;
  int r = row0 + (lane & 15);
  if (r >= N) r = N - 1;        // garbage only pollutes unstored D rows
  long roff = (long)r * 128;
  float dscale = deg[r];
  v8f c = {};
#pragma unroll
  for (int kt = 0; kt < 4; kt++)
    c = WMMA_BF16(a_frag_f32(h + roff, kt, lane, dscale),
                  b_frag(pWv, 8, kt, nt, lane), c);
#pragma unroll
  for (int kt = 0; kt < 4; kt++)
    c = WMMA_BF16(a_frag_f32(agg + roff, kt, lane, 1.f),
                  b_frag(pWw, 8, kt, nt, lane), c);
#pragma unroll
  for (int kt = 0; kt < 4; kt++)
    c = WMMA_BF16(a_frag_f32(ehat + roff, kt, lane, 1.f),
                  b_frag(pWe, 8, kt, nt, lane), c);
  int col = (lane & 15) + (nt << 4);
  float cv = cond[col];
#pragma unroll
  for (int v = 0; v < 8; v++) {
    int row = row0 + v + ((lane >> 4) << 3);   // C layout: M = v + 8*(lane>=16)
    if (row < N) m[(long)row * 128 + col] = c[v] + deg[row] * cv;
  }
}

// GRU cell, in-place on h. gi = m@Wih+bih, gh = h@Whh+bhh (384 cols = 24 tiles;
// wave w owns tiles {w, w+8, w+16} = {r,z,n} slices of the same 16 columns).
__global__ void __launch_bounds__(256) k_gru(
    float* __restrict__ h, const float* __restrict__ m,
    const unsigned short* __restrict__ pWih,
    const unsigned short* __restrict__ pWhh,
    const float* __restrict__ bih, const float* __restrict__ bhh, int N) {
  int lane = threadIdx.x & 31;
  int w    = threadIdx.x >> 5;
  int row0 = blockIdx.x << 4;
  int r = row0 + (lane & 15);
  if (r >= N) r = N - 1;
  long roff = (long)r * 128;
  v8f gi0 = {}, gi1 = {}, gi2 = {}, gh0 = {}, gh1 = {}, gh2 = {};
#pragma unroll
  for (int kt = 0; kt < 4; kt++) {
    v16bf am = a_frag_f32(m + roff, kt, lane, 1.f);
    v16bf ah = a_frag_f32(h + roff, kt, lane, 1.f);
    gi0 = WMMA_BF16(am, b_frag(pWih, 24, kt, w,      lane), gi0);
    gi1 = WMMA_BF16(am, b_frag(pWih, 24, kt, w + 8,  lane), gi1);
    gi2 = WMMA_BF16(am, b_frag(pWih, 24, kt, w + 16, lane), gi2);
    gh0 = WMMA_BF16(ah, b_frag(pWhh, 24, kt, w,      lane), gh0);
    gh1 = WMMA_BF16(ah, b_frag(pWhh, 24, kt, w + 8,  lane), gh1);
    gh2 = WMMA_BF16(ah, b_frag(pWhh, 24, kt, w + 16, lane), gh2);
  }
  int col = (lane & 15) + (w << 4);
  float bir = bih[col], biz = bih[128 + col], bin = bih[256 + col];
  float bhr = bhh[col], bhz = bhh[128 + col], bhn = bhh[256 + col];
  __syncthreads();   // all waves finished reading h rows before in-place write
#pragma unroll
  for (int v = 0; v < 8; v++) {
    int row = row0 + v + ((lane >> 4) << 3);
    if (row >= N) continue;
    float ir = gi0[v] + bir, iz = gi1[v] + biz, inn = gi2[v] + bin;
    float hr = gh0[v] + bhr, hz = gh1[v] + bhz, hn = gh2[v] + bhn;
    float rg = 1.f / (1.f + __expf(-(ir + hr)));
    float zg = 1.f / (1.f + __expf(-(iz + hz)));
    float ng = tanhf(inn + rg * hn);
    long idx = (long)row * 128 + col;
    float hold = h[idx];
    h[idx] = (1.f - zg) * ng + zg * hold;
  }
}

// gated = (h@W1+b1)*sigmoid(h@W2+b2); gsum[col] += sum over the block's rows.
__global__ void __launch_bounds__(256) k_gated_mean(
    const float* __restrict__ h,
    const unsigned short* __restrict__ pW1,
    const unsigned short* __restrict__ pW2,
    const float* __restrict__ b1, const float* __restrict__ b2,
    float* __restrict__ gsum, int N) {
  int lane = threadIdx.x & 31;
  int nt   = threadIdx.x >> 5;
  int row0 = blockIdx.x << 4;
  int r = row0 + (lane & 15);
  if (r >= N) r = N - 1;
  long roff = (long)r * 128;
  v8f c1 = {}, c2 = {};
#pragma unroll
  for (int kt = 0; kt < 4; kt++) {
    v16bf a = a_frag_f32(h + roff, kt, lane, 1.f);
    c1 = WMMA_BF16(a, b_frag(pW1, 8, kt, nt, lane), c1);
    c2 = WMMA_BF16(a, b_frag(pW2, 8, kt, nt, lane), c2);
  }
  int col = (lane & 15) + (nt << 4);
  float bb1 = b1[col], bb2 = b2[col];
  float s = 0.f;
#pragma unroll
  for (int v = 0; v < 8; v++) {
    int row = row0 + v + ((lane >> 4) << 3);
    if (row < N)
      s += (c1[v] + bb1) * (1.f / (1.f + __expf(-(c2[v] + bb2))));
  }
  s += __shfl_xor(s, 16, 32);          // lane <-> lane+16 share a column (wave32)
  if (lane < 16) atomic_add_f32(gsum + col, s);
}

// VAE head on the pooled 1x128 vector; out = [latent, mu, logvar] (1x384).
__global__ void k_finalize(const float* __restrict__ gsum,
                           const float* __restrict__ mean_W,
                           const float* __restrict__ mean_b,
                           const float* __restrict__ logvar_W,
                           const float* __restrict__ logvar_b,
                           const float* __restrict__ eps,
                           float* __restrict__ out, float invN) {
  int c = threadIdx.x;
  __shared__ float enc[128];
  enc[c] = gsum[c] * invN;
  __syncthreads();
  float mu = mean_b[c], lv = logvar_b[c];
#pragma unroll 8
  for (int k = 0; k < 128; k++) {
    float e = enc[k];
    mu += e * mean_W[k * 128 + c];
    lv += e * logvar_W[k * 128 + c];
  }
  float lat = mu + eps[c] * __expf(0.5f * lv);
  out[c] = lat;
  out[128 + c] = mu;
  out[256 + c] = lv;
}

// ---------------------------------------------------------------------------
extern "C" void kernel_launch(void* const* d_in, const int* in_sizes, int n_in,
                              void* d_out, int out_size, void* d_ws, size_t ws_size,
                              hipStream_t stream) {
  const float* node_feats = (const float*)d_in[0];
  const float* edge_feats = (const float*)d_in[1];
  const float* condition  = (const float*)d_in[2];
  const float* eps        = (const float*)d_in[3];
  const float* W_node_emb = (const float*)d_in[4];
  const float* W_edge_emb = (const float*)d_in[5];
  const float* enc_U_W    = (const float*)d_in[6];
  const float* enc_U_b    = (const float*)d_in[7];
  const float* gru_Wih    = (const float*)d_in[8];
  const float* gru_Whh    = (const float*)d_in[9];
  const float* gru_bih    = (const float*)d_in[10];
  const float* gru_bhh    = (const float*)d_in[11];
  const float* cal1_W     = (const float*)d_in[12];
  const float* cal1_b     = (const float*)d_in[13];
  const float* cal2_W     = (const float*)d_in[14];
  const float* cal2_b     = (const float*)d_in[15];
  const float* mean_W     = (const float*)d_in[16];
  const float* mean_b     = (const float*)d_in[17];
  const float* logvar_W   = (const float*)d_in[18];
  const float* logvar_b   = (const float*)d_in[19];
  const int*   edge_src   = (const int*)d_in[20];
  const int*   edge_dst   = (const int*)d_in[21];

  const long N = in_sizes[0] / 14;   // 50000
  const long E = in_sizes[1] / 11;   // 500000

  // workspace carve-out (256B-aligned sub-buffers), ~104 MB total
  char* wp = (char*)d_ws;
  auto alloc = [&](size_t bytes) -> void* {
    void* p = (void*)wp;
    wp += (bytes + 255) & ~(size_t)255;
    return p;
  };
  float* h    = (float*)alloc((size_t)N * 128 * sizeof(float));
  float* m    = (float*)alloc((size_t)N * 128 * sizeof(float));
  float* agg  = (float*)alloc((size_t)N * 128 * sizeof(float));
  float* ehat = (float*)alloc((size_t)N * 128 * sizeof(float));
  float* deg  = (float*)alloc((size_t)N * sizeof(float));
  float* gsum = (float*)alloc(128 * sizeof(float));
  float* condt= (float*)alloc(3 * 128 * sizeof(float));
  unsigned short *pWv[3], *pWw[3], *pWe[3], *pWih[3], *pWhh[3];
  for (int k = 0; k < 3; k++) {
    pWv[k]  = (unsigned short*)alloc(128 * 128 * 2);
    pWw[k]  = (unsigned short*)alloc(128 * 128 * 2);
    pWe[k]  = (unsigned short*)alloc(128 * 128 * 2);
    pWih[k] = (unsigned short*)alloc(128 * 384 * 2);
    pWhh[k] = (unsigned short*)alloc(128 * 384 * 2);
  }
  unsigned short* pW1 = (unsigned short*)alloc(128 * 128 * 2);
  unsigned short* pW2 = (unsigned short*)alloc(128 * 128 * 2);

  const int BT = 256;
  const int nBlkNE  = (int)((N * 128 + BT - 1) / BT);   // N*128 threads
  const int nBlkE32 = (int)((E * 32  + BT - 1) / BT);   // E*32 threads
  const int nBlkRow = (int)((N + 15) / 16);             // 16-row WMMA tiles
  const int nBlkN   = (int)((N + BT - 1) / BT);

  // zero accumulators (rebuilt every call — harness does not re-poison)
  k_zero<<<nBlkNE, BT, 0, stream>>>(ehat, N * 128);
  k_zero<<<nBlkN,  BT, 0, stream>>>(deg, N);
  k_zero<<<1,      BT, 0, stream>>>(gsum, 128);

  // pack weights to bf16 WMMA fragments
  for (int k = 0; k < 3; k++) {
    const float* Uk = enc_U_W + (long)k * 388 * 128;
    k_prepack<<<64,  BT, 0, stream>>>(Uk,             pWv[k], 128);
    k_prepack<<<64,  BT, 0, stream>>>(Uk + 128 * 128, pWw[k], 128);
    k_prepack<<<64,  BT, 0, stream>>>(Uk + 256 * 128, pWe[k], 128);
    k_prepack<<<192, BT, 0, stream>>>(gru_Wih + (long)k * 128 * 384, pWih[k], 384);
    k_prepack<<<192, BT, 0, stream>>>(gru_Whh + (long)k * 128 * 384, pWhh[k], 384);
  }
  k_prepack<<<64, BT, 0, stream>>>(cal1_W, pW1, 128);
  k_prepack<<<64, BT, 0, stream>>>(cal2_W, pW2, 128);

  k_cond<<<3, 128, 0, stream>>>(condition, enc_U_W, enc_U_b, condt);
  k_node_embed<<<nBlkNE, BT, 0, stream>>>(node_feats, W_node_emb, h, N);
  k_edge_embed_scatter<<<nBlkE32, BT, 0, stream>>>(edge_feats, W_edge_emb,
                                                   edge_dst, ehat, deg, E);

  for (int k = 0; k < 3; k++) {
    k_zero<<<nBlkNE, BT, 0, stream>>>(agg, N * 128);
    k_scatter_h<<<nBlkE32, BT, 0, stream>>>(h, edge_src, edge_dst, agg, E);
    k_gemm_msg<<<nBlkRow, BT, 0, stream>>>(h, agg, ehat, deg, condt + k * 128,
                                           pWv[k], pWw[k], pWe[k], m, (int)N);
    k_gru<<<nBlkRow, BT, 0, stream>>>(h, m, pWih[k], pWhh[k],
                                      gru_bih + k * 384, gru_bhh + k * 384, (int)N);
  }

  k_gated_mean<<<nBlkRow, BT, 0, stream>>>(h, pW1, pW2, cal1_b, cal2_b, gsum, (int)N);
  k_finalize<<<1, 128, 0, stream>>>(gsum, mean_W, mean_b, logvar_W, logvar_b,
                                    eps, (float*)d_out, 1.0f / (float)N);
}
